// YOLOv5_17136919511508
// MI455X (gfx1250) — matrix operand order repaired and checked by reading, not compiled, not run on hardware
//
#include <hip/hip_runtime.h>
#include <hip/hip_fp16.h>

// ---------------------------------------------------------------------------
// YOLOv5 decode + top-k + NMS + IoU-weighted merge for MI455X (gfx1250).
//
//   K1 decode_hist : stream 137MB preds once; decode+clip boxes -> ws;
//                    per-image 2048-bin score histogram via LDS atomics.
//   K2 findcut     : per image, suffix-scan histogram for >=300 cutoff bin.
//   K3 compact     : stream preds again; append (score, flat_idx) >= cutoff.
//   K4 nms_merge   : one 256-thread block / image, all in LDS:
//                    TDM tensor_load_to_lds of candidate arrays ->
//                    bitonic sort -> top300 -> sequential NMS scan ->
//                    IoU-weighted merge as [112x320]x[320x16] WMMA matmul
//                    with fragment-order LDS layout (2x ds_load_b128/frag).
// ---------------------------------------------------------------------------

typedef __attribute__((ext_vector_type(16))) _Float16 v16h;
typedef __attribute__((ext_vector_type(8)))  float    v8f;
typedef __attribute__((ext_vector_type(4)))  unsigned v4u;
typedef __attribute__((ext_vector_type(8)))  unsigned v8u;

#define NUM_CLASSES 80
#define CT          25200           // 80*80*3 + 40*40*3 + 20*20*3
#define NIMG        16
#define TOPK        300
#define KPAD        320             // TOPK padded to WMMA K multiple
#define KT_N        (KPAD / 32)     // 10 K-tiles
#define DET         100
#define MPAD        112             // DET padded to 16
#define HBINS       2048
#define CAND_CAP    2048
#define SCORE_T     0.1f
#define NMS_T       0.6f
#define BIN_SCALE   (HBINS / 0.9f)  // scores in (0.1, 1.0]

__constant__ float c_anch[3][3][2] = {
    {{6.1f, 8.1f}, {20.6f, 12.6f}, {11.2f, 23.7f}},
    {{36.2f, 26.8f}, {25.9f, 57.2f}, {57.8f, 47.9f}},
    {{122.1f, 78.3f}, {73.7f, 143.8f}, {236.1f, 213.1f}}};

__device__ __forceinline__ float sigf(float x) {
    return 1.0f / (1.0f + __expf(-x));
}

__device__ __forceinline__ const float* anchor_src(
    int n, int a, const float* p0, const float* p1, const float* p2,
    int& lvl, int& ai, int& gx, int& gy, float& stride) {
    int cell, W;
    const float* src;
    if (a < 19200)      { cell = a;         src = p0 + ((size_t)n * 19200 + cell) * 85; W = 80; stride =  8.f; lvl = 0; }
    else if (a < 24000) { cell = a - 19200; src = p1 + ((size_t)n *  4800 + cell) * 85; W = 40; stride = 16.f; lvl = 1; }
    else                { cell = a - 24000; src = p2 + ((size_t)n *  1200 + cell) * 85; W = 20; stride = 32.f; lvl = 2; }
    ai = cell % 3;
    int gi = cell / 3;
    gx = gi % W;
    gy = gi / W;
    return src;
}

__device__ __forceinline__ float iou_xyxy(float ax1, float ay1, float ax2, float ay2,
                                          float bx1, float by1, float bx2, float by2) {
    float areaA = (ax2 - ax1) * (ay2 - ay1);
    float areaB = (bx2 - bx1) * (by2 - by1);
    float lx = fmaxf(ax1, bx1), ly = fmaxf(ay1, by1);
    float rx = fminf(ax2, bx2), ry = fminf(ay2, by2);
    float iw = fmaxf(rx - lx, 0.f), ih = fmaxf(ry - ly, 0.f);
    float inter = iw * ih;
    return inter / (areaA + areaB - inter + 1e-7f);
}

// Build a 1-D TDM descriptor (D#) for 2048 x 4B global -> LDS and issue it.
// group0: [1:0]=count=1, [63:32]=lds_addr, [120:64]=global_addr, [127:126]=type 2
// group1: [17:16]=data_size(4B)=2, [79:48]=tensor_dim0, [111:80]=tensor_dim1=1,
//         [127:112]=tile_dim0, [207:160]=tensor_dim0_stride
__device__ __forceinline__ void tdm_load_1d_2048x4(unsigned lds_addr,
                                                   unsigned long long gaddr) {
    v4u g0;
    g0[0] = 1u;                                     // count=1, gather off
    g0[1] = lds_addr;                               // LDS byte address
    g0[2] = (unsigned)gaddr;                        // global addr [31:0]
    g0[3] = ((unsigned)(gaddr >> 32) & 0x01FFFFFFu) // global addr [56:32]
            | (2u << 30);                           // type = 2 (image)
    v8u g1;
    g1[0] = 2u << 16;                               // data_size = 4B
    g1[1] = (unsigned)CAND_CAP << 16;               // tensor_dim0 lo16 @ [63:48]
    g1[2] = 1u << 16;                               // tensor_dim1 = 1 @ [95:80]
    g1[3] = (unsigned)CAND_CAP << 16;               // tile_dim0 @ [127:112]
    g1[4] = 0u;                                     // tile_dim1/2 unused
    g1[5] = (unsigned)CAND_CAP;                     // tensor_dim0_stride lo32
    g1[6] = 0u;
    g1[7] = 0u;
    asm volatile("tensor_load_to_lds %0, %1" :: "s"(g0), "s"(g1) : "memory");
}

// ---------------- K1: decode boxes + per-image score histogram --------------
__global__ __launch_bounds__(256) void yolo_decode_hist(
    const float* __restrict__ p0, const float* __restrict__ p1,
    const float* __restrict__ p2, float* __restrict__ boxes,
    unsigned* __restrict__ hist) {
    __shared__ unsigned lh[HBINS];
    const int n = blockIdx.y;
    for (int b = threadIdx.x; b < HBINS; b += 256) lh[b] = 0u;
    __syncthreads();

    const int a = blockIdx.x * 256 + threadIdx.x;
    if (a < CT) {
        int lvl, ai, gx, gy; float stride;
        const float* src = anchor_src(n, a, p0, p1, p2, lvl, ai, gx, gy, stride);
        __builtin_prefetch(src + 85, 0, 1);   // global_prefetch_b8 for next row

        float px = sigf(src[0]), py = sigf(src[1]);
        float pw = sigf(src[2]), ph = sigf(src[3]);
        float obj = sigf(src[4]);
        float cx = (2.f * px - 0.5f + (float)gx) * stride;
        float cy = (2.f * py - 0.5f + (float)gy) * stride;
        float w  = 4.f * pw * pw * c_anch[lvl][ai][0];
        float h  = 4.f * ph * ph * c_anch[lvl][ai][1];
        float4 bx;
        bx.x = fminf(fmaxf(cx - 0.5f * w, 0.f), 640.f);
        bx.y = fminf(fmaxf(cy - 0.5f * h, 0.f), 640.f);
        bx.z = fminf(fmaxf(cx + 0.5f * w, 0.f), 640.f);
        bx.w = fminf(fmaxf(cy + 0.5f * h, 0.f), 640.f);
        *(float4*)&boxes[((size_t)n * CT + a) * 4] = bx;

        if (obj > SCORE_T) {
            for (int c = 0; c < NUM_CLASSES; ++c) {
                float s = obj * sigf(src[5 + c]);
                if (s > SCORE_T) {
                    int b = (int)((s - SCORE_T) * BIN_SCALE);
                    b = b < 0 ? 0 : (b > HBINS - 1 ? HBINS - 1 : b);
                    atomicAdd(&lh[b], 1u);
                }
            }
        }
    }
    __syncthreads();
    unsigned* gh = hist + (size_t)n * HBINS;
    for (int b = threadIdx.x; b < HBINS; b += 256) {
        unsigned v = lh[b];
        if (v) atomicAdd(&gh[b], v);
    }
}

// ---------------- K2: per-image cutoff bin (>= TOPK candidates) -------------
__global__ void yolo_findcut(const unsigned* __restrict__ hist, int* __restrict__ cut) {
    if (threadIdx.x == 0) {
        const unsigned* h = hist + (size_t)blockIdx.x * HBINS;
        unsigned cum = 0;
        int c = 0;
        for (int b = HBINS - 1; b >= 0; --b) {
            cum += h[b];
            if (cum >= TOPK) { c = b; break; }
        }
        cut[blockIdx.x] = c;
    }
}

// ---------------- K3: compact candidates >= cutoff --------------------------
__global__ __launch_bounds__(256) void yolo_compact(
    const float* __restrict__ p0, const float* __restrict__ p1,
    const float* __restrict__ p2, const int* __restrict__ cut,
    float* __restrict__ cscore, int* __restrict__ cidx,
    int* __restrict__ ccnt) {
    const int n = blockIdx.y;
    const int a = blockIdx.x * 256 + threadIdx.x;
    if (a >= CT) return;
    int lvl, ai, gx, gy; float stride;
    const float* src = anchor_src(n, a, p0, p1, p2, lvl, ai, gx, gy, stride);
    __builtin_prefetch(src + 85, 0, 1);
    float obj = sigf(src[4]);
    if (obj <= SCORE_T) return;
    const int mycut = cut[n];
    for (int c = 0; c < NUM_CLASSES; ++c) {
        float s = obj * sigf(src[5 + c]);
        if (s > SCORE_T) {
            int b = (int)((s - SCORE_T) * BIN_SCALE);
            b = b < 0 ? 0 : (b > HBINS - 1 ? HBINS - 1 : b);
            if (b >= mycut) {
                int slot = atomicAdd(&ccnt[n], 1);
                if (slot < CAND_CAP) {
                    cscore[(size_t)n * CAND_CAP + slot] = s;
                    cidx[(size_t)n * CAND_CAP + slot]   = a * NUM_CLASSES + c;
                }
            }
        }
    }
}

// ---------------- K4: sort + NMS + WMMA merge, one block per image ----------
// Dynamic LDS layout (bytes):
#define L_SC   0                       // float [2048]
#define L_SI   8192                    // int   [2048]
#define L_CB   16384                   // float [320*4]
#define L_CLAB 21504                   // int   [320]
#define L_CSC  22784                   // float [320]
#define L_SUP  24064                   // int   [320]
#define L_DB   25344                   // float [112*4]
#define L_DLAB 27136                   // int   [112]
#define L_DSC  27584                   // float [112]
#define L_SEL  28032                   // int   [100]
#define L_MISC 28432                   // int   [few]
#define L_W    28544                   // f16   [7*10*32*16] fragment order
#define L_B    100224                  // f16   [10*32*16]   fragment order
#define L_D    110464                  // float [112*16]
#define L_TOTAL 117632

__global__ __launch_bounds__(256) void yolo_nms_merge(
    const float* __restrict__ boxes, const float* __restrict__ cscore,
    const int* __restrict__ cidx, const int* __restrict__ ccnt,
    float* __restrict__ out) {
    extern __shared__ unsigned char smem[];
    float*    sc   = (float*)(smem + L_SC);
    int*      si   = (int*)(smem + L_SI);
    float*    cb   = (float*)(smem + L_CB);
    int*      clab = (int*)(smem + L_CLAB);
    float*    csc  = (float*)(smem + L_CSC);
    int*      sup  = (int*)(smem + L_SUP);
    float*    db   = (float*)(smem + L_DB);
    int*      dlab = (int*)(smem + L_DLAB);
    float*    dsc  = (float*)(smem + L_DSC);
    int*      sel  = (int*)(smem + L_SEL);
    int*      misc = (int*)(smem + L_MISC);
    _Float16* Wm   = (_Float16*)(smem + L_W);
    _Float16* Bm   = (_Float16*)(smem + L_B);
    float*    Dm   = (float*)(smem + L_D);

    const int n = blockIdx.x;
    const int tid = threadIdx.x;
    const int wv = tid >> 5;
    const int lane = tid & 31;
    int total = ccnt[n];
    if (total > CAND_CAP) total = CAND_CAP;

    // ---- TDM: DMA both candidate arrays into LDS (wave 0 issues) ----
    if (wv == 0) {
        tdm_load_1d_2048x4((unsigned)(unsigned long long)(uintptr_t)sc,
                           (unsigned long long)(uintptr_t)(cscore + (size_t)n * CAND_CAP));
        tdm_load_1d_2048x4((unsigned)(unsigned long long)(uintptr_t)si,
                           (unsigned long long)(uintptr_t)(cidx + (size_t)n * CAND_CAP));
        __builtin_amdgcn_s_wait_tensorcnt(0);
    }
    __syncthreads();
    // patch slots beyond the valid count with sentinel values
    for (int t = tid; t < CAND_CAP; t += 256) {
        if (t >= total) { sc[t] = -1.f; si[t] = 0; }
    }
    __syncthreads();

    // ---- bitonic sort, descending by score ----
    for (int k = 2; k <= CAND_CAP; k <<= 1) {
        for (int j = k >> 1; j > 0; j >>= 1) {
            for (int t = tid; t < CAND_CAP; t += 256) {
                int ixj = t ^ j;
                if (ixj > t) {
                    bool dirDesc = ((t & k) == 0);
                    float A = sc[t], B = sc[ixj];
                    bool sw = dirDesc ? (A < B) : (A > B);
                    if (sw) {
                        sc[t] = B; sc[ixj] = A;
                        int ti = si[t]; si[t] = si[ixj]; si[ixj] = ti;
                    }
                }
            }
            __syncthreads();
        }
    }

    // ---- gather top-300 boxes/labels (entries 300..319 padded invalid) ----
    for (int t = tid; t < KPAD; t += 256) {
        float s = (t < TOPK) ? sc[t] : 0.f;
        bool v = (t < TOPK) && (t < total) && (s > SCORE_T);
        if (v) {
            int fi = si[t];
            int cand = fi / NUM_CLASSES;
            clab[t] = fi - cand * NUM_CLASSES;
            float4 bx = *(const float4*)&boxes[((size_t)n * CT + cand) * 4];
            cb[t * 4 + 0] = bx.x; cb[t * 4 + 1] = bx.y;
            cb[t * 4 + 2] = bx.z; cb[t * 4 + 3] = bx.w;
            csc[t] = s;
            sup[t] = 0;
        } else {
            cb[t * 4 + 0] = 0.f; cb[t * 4 + 1] = 0.f;
            cb[t * 4 + 2] = 0.f; cb[t * 4 + 3] = 0.f;
            clab[t] = -1;
            csc[t] = 0.f;
            sup[t] = 1;
        }
    }
    __syncthreads();

    // ---- sequential NMS scan (same-class == class-offset trick) ----
    for (int i = 0; i < TOPK; ++i) {
        if (sup[i] == 0) {
            float ax1 = cb[i * 4 + 0], ay1 = cb[i * 4 + 1];
            float ax2 = cb[i * 4 + 2], ay2 = cb[i * 4 + 3];
            int li = clab[i];
            for (int j = i + 1 + tid; j < TOPK; j += 256) {
                if (clab[j] == li) {
                    float v = iou_xyxy(ax1, ay1, ax2, ay2,
                                       cb[j * 4 + 0], cb[j * 4 + 1],
                                       cb[j * 4 + 2], cb[j * 4 + 3]);
                    if (v > NMS_T) sup[j] = 1;
                }
            }
        }
        __syncthreads();
    }

    // ---- select first 100 kept (score order) ----
    if (tid == 0) {
        int c = 0;
        for (int i = 0; i < TOPK; ++i) {
            if (!sup[i]) {
                if (c < DET) sel[c] = i;
                ++c;
            }
        }
        misc[0] = (c < DET) ? c : DET;
    }
    __syncthreads();
    const int ndet = misc[0];
    for (int d = tid; d < MPAD; d += 256) {
        if (d < ndet) {
            int j = sel[d];
            db[d * 4 + 0] = cb[j * 4 + 0]; db[d * 4 + 1] = cb[j * 4 + 1];
            db[d * 4 + 2] = cb[j * 4 + 2]; db[d * 4 + 3] = cb[j * 4 + 3];
            dlab[d] = clab[j];
            dsc[d]  = csc[j];
        } else {
            db[d * 4 + 0] = 0.f; db[d * 4 + 1] = 0.f;
            db[d * 4 + 2] = 0.f; db[d * 4 + 3] = 0.f;
            dlab[d] = -1;
            dsc[d]  = 0.f;
        }
    }
    __syncthreads();

    // ---- build W [112x320] and B [320x16] in f16, FRAGMENT ORDER ----
    // A-frag (16-bit 16x32, ISA 7.12.2): lane = lg*16 + rowInTile, where
    // lg=0 covers K {0..7,16..23}, lg=1 covers K {8..15,24..31}; half index
    // h = 2*v2 + (kk&1) with v2 per the VGPR table. Each lane's 16 halfs are
    // stored contiguously -> v16h load = 2x ds_load_b128.
    for (int e = tid; e < MPAD * KPAD; e += 256) {
        int i = e / KPAD, j = e - i * KPAD;
        float w = 0.f;
        int li = dlab[i];
        if (li >= 0 && clab[j] == li) {
            float v = iou_xyxy(db[i * 4 + 0], db[i * 4 + 1], db[i * 4 + 2], db[i * 4 + 3],
                               cb[j * 4 + 0], cb[j * 4 + 1], cb[j * 4 + 2], cb[j * 4 + 3]);
            if (v > NMS_T) w = csc[j];
        }
        int tv = i >> 4, r = i & 15;
        int kt = j >> 5, kk = j & 31;
        int lg = (kk >> 3) & 1;
        int v2 = (kk < 16) ? ((kk & 7) >> 1) : (4 + ((kk & 7) >> 1));
        int h  = 2 * v2 + (kk & 1);
        int ln = lg * 16 + r;
        Wm[((tv * KT_N + kt) * 32 + ln) * 16 + h] = (_Float16)w;
    }
    // B-frag (32x16): lane = lg*16 + col, lg=0 covers K {0..7,16..23},
    // lg=1 covers K {8..15,24..31} (same as A); contiguous per lane.
    for (int e = tid; e < KPAD * 16; e += 256) {
        int k = e / 16, c = e - k * 16;
        float v = (c < 4) ? cb[k * 4 + c] : ((c == 4) ? 1.f : 0.f);
        int kt = k >> 5, kk = k & 31;
        int lg = (kk >> 3) & 1;
        int h  = (kk & 7) + ((kk >> 4) << 3);
        int ln = lg * 16 + c;
        Bm[(kt * 32 + ln) * 16 + h] = (_Float16)v;
    }
    __syncthreads();

    // ---- WMMA: D[112x16] = W[112x320] x B[320x16], one M-tile per wave ----
    if (wv < MPAD / 16) {
        v8f acc;
#pragma unroll
        for (int r = 0; r < 8; ++r) acc[r] = 0.f;
        const int colB = lane & 15;
#pragma unroll
        for (int kt = 0; kt < KT_N; ++kt) {
            v16h Af = *(const v16h*)&Wm[((wv * KT_N + kt) * 32 + lane) * 16];
            v16h Bf = *(const v16h*)&Bm[(kt * 32 + lane) * 16];
            acc = __builtin_amdgcn_wmma_f32_16x16x32_f16(
                false, Af, false, Bf, (short)0, acc, false, false);
        }
#pragma unroll
        for (int r = 0; r < 8; ++r) {
            int row = wv * 16 + r + ((lane < 16) ? 0 : 8);
            Dm[row * 16 + colB] = acc[r];
        }
    }
    __syncthreads();

    // ---- final: normalize merged boxes, emit outputs ----
    float* obox = out;                                   // [16,100,4] f32
    int*   olab = (int*)out + NIMG * DET * 4;            // [16,100] i32
    float* oscr = out + NIMG * DET * 4 + NIMG * DET;     // [16,100] f32
    for (int d = tid; d < DET; d += 256) {
        bool dv = d < ndet;
        float wsum = Dm[d * 16 + 4];
        float inv  = 1.f / fmaxf(wsum, 1e-7f);
        bool useM  = dv && (wsum > 0.f);
        for (int c = 0; c < 4; ++c) {
            float bx = useM ? Dm[d * 16 + c] * inv : db[d * 4 + c];
            obox[((size_t)n * DET + d) * 4 + c] = bx;
        }
        olab[n * DET + d] = dlab[d];
        oscr[n * DET + d] = dsc[d];
    }
}

// ---------------------------------------------------------------------------
extern "C" void kernel_launch(void* const* d_in, const int* in_sizes, int n_in,
                              void* d_out, int out_size, void* d_ws, size_t ws_size,
                              hipStream_t stream) {
    const float* p0 = (const float*)d_in[0];   // [16,80,80,3,85]
    const float* p1 = (const float*)d_in[1];   // [16,40,40,3,85]
    const float* p2 = (const float*)d_in[2];   // [16,20,20,3,85]

    char* ws = (char*)d_ws;
    float*    boxes = (float*)ws;                                  // 6,451,200 B
    unsigned* hist  = (unsigned*)(ws + 6451200);                   //   131,072 B
    int*      cut   = (int*)(ws + 6451200 + 131072);               //        64 B
    int*      ccnt  = (int*)(ws + 6451200 + 131072 + 64);          //        64 B
    float*    csv   = (float*)(ws + 6451200 + 131072 + 128);       //   131,072 B
    int*      civ   = (int*)(ws + 6451200 + 131072 + 128 + 131072);//   131,072 B

    // zero hist + cut + ccnt each call (graph-capturable memset node)
    hipMemsetAsync(hist, 0, 131072 + 128, stream);

    dim3 gdec((CT + 255) / 256, NIMG);
    yolo_decode_hist<<<gdec, 256, 0, stream>>>(p0, p1, p2, boxes, hist);
    yolo_findcut<<<NIMG, 32, 0, stream>>>(hist, cut);
    yolo_compact<<<gdec, 256, 0, stream>>>(p0, p1, p2, cut, csv, civ, ccnt);
    yolo_nms_merge<<<NIMG, 256, L_TOTAL, stream>>>(boxes, csv, civ, ccnt,
                                                   (float*)d_out);
}
